// HyperbolicMessagePassing_33818572489034
// MI455X (gfx1250) — matrix operand order repaired and checked by reading, not compiled, not run on hardware
//
#include <hip/hip_runtime.h>
#include <hip/hip_bf16.h>
#include <math.h>

// HyperbolicMessagePassing on the Poincare ball (c=1), N=1536, C=64.
// log_ij = A_ij*x_i + B_ij*x_j  (scalars from Gram matrix) =>
//   agg = diag(rowsum(W.*A)) * x + (W.*B) @ x
// Two big GEMMs run on v_wmma_f32_16x16x32_f16 (wave32, 16x16 tile per wave).

typedef __attribute__((ext_vector_type(16))) _Float16 v16h;
typedef __attribute__((ext_vector_type(8)))  _Float16 v8h;
typedef __attribute__((ext_vector_type(8)))  float    v8f;

#define NN   1536
#define CC   64
#define NT   96          // NN/16 tiles per dimension
#define MIN_NORM   1e-15f
#define TANH_CLIP  (1.0f - 1e-5f)

// ---- WMMA tile loaders ------------------------------------------------------
// A-matrix 16x32 f16 tile from a row-major [*, ld] source at (row0, k0).
// Layout: lane = 16*half + m holds row (row0+m);
//   element e -> K = 16*half + e (one contiguous 32-byte load, matches B)
__device__ __forceinline__ v16h load_a16(const _Float16* __restrict__ src,
                                         int ld, int row0, int k0,
                                         int m, int half) {
  const _Float16* p = src + (row0 + m) * ld + k0;
  return *(const v16h*)(p + half * 16);
}

// B-matrix 32x16 f16 tile, where column n of B is row (col0+n) of the
// row-major source (i.e. source holds B transposed). Lane = 16*half + m holds
// column (col0+m); element e -> K = 16*half + e (one contiguous 32-byte load).
__device__ __forceinline__ v16h load_b16(const _Float16* __restrict__ src,
                                         int ld, int col0, int k0,
                                         int m, int half) {
  return *(const v16h*)(src + (col0 + m) * ld + k0 + half * 16);
}

// ---- K0: prep: f16 copies, row norms, degree inverse, zero accumulators ----
__global__ void k_prep(const float* __restrict__ x, const float* __restrict__ adj,
                       _Float16* __restrict__ xh, _Float16* __restrict__ xhT,
                       float* __restrict__ n2, float* __restrict__ dinv,
                       float* __restrict__ rowA) {
  const int i    = (blockIdx.x * blockDim.x + threadIdx.x) >> 5;  // row 0..1535
  const int lane = threadIdx.x & 31;

  const float a0 = x[i * CC + lane];
  const float a1 = x[i * CC + lane + 32];
  xh[i * CC + lane]        = (_Float16)a0;
  xh[i * CC + lane + 32]   = (_Float16)a1;
  xhT[lane * NN + i]        = (_Float16)a0;
  xhT[(lane + 32) * NN + i] = (_Float16)a1;

  float s = a0 * a0 + a1 * a1;
#pragma unroll
  for (int mm = 16; mm >= 1; mm >>= 1) s += __shfl_xor(s, mm, 32);

  float dsum = 0.0f;
  for (int j = lane; j < NN; j += 32) dsum += adj[i * NN + j];
#pragma unroll
  for (int mm = 16; mm >= 1; mm >>= 1) dsum += __shfl_xor(dsum, mm, 32);

  if (lane == 0) {
    n2[i]   = s;
    dinv[i] = 1.0f / fmaxf(dsum, 1e-8f);
    rowA[i] = 0.0f;
  }
}

// ---- K1: Gram tile via WMMA + pairwise scalar field -------------------------
// One wave per 16x16 tile of G = x @ x^T; epilogue computes A_ij/B_ij,
// writes Mh = (W.*B) in f16, reduces W.*A over j into rowA via atomics.
__global__ void k_pair(const float* __restrict__ adj,
                       const _Float16* __restrict__ xh,
                       const float* __restrict__ n2,
                       const float* __restrict__ dinv,
                       _Float16* __restrict__ Mh,
                       float* __restrict__ rowA) {
  const int wave = (blockIdx.x * blockDim.x + threadIdx.x) >> 5;
  const int ti = wave / NT, tj = wave % NT;
  const int i0 = ti * 16, j0 = tj * 16;
  const int lane = threadIdx.x & 31;
  const int m = lane & 15, half = lane >> 4;

  v8f d = {};
#pragma unroll
  for (int k0 = 0; k0 < CC; k0 += 32) {
    v16h a = load_a16(xh, CC, i0, k0, m, half);
    v16h b = load_b16(xh, CC, j0, k0, m, half);
    d = __builtin_amdgcn_wmma_f32_16x16x32_f16(false, a, false, b,
                                               (short)0, d, false, false);
  }

  const int   j   = j0 + m;
  const float n2j = n2[j];
#pragma unroll
  for (int r = 0; r < 8; ++r) {
    const int   i   = i0 + r + 8 * half;      // D layout: VGPR r -> M = r+8*half
    const float n2i = n2[i];
    const float g   = d[r];                   // x_i . x_j
    const float den = fmaxf(1.0f - 2.0f * g + n2i * n2j, MIN_NORM);
    const float p   = -(1.0f - 2.0f * g + n2j) / den;
    const float q   = (1.0f - n2i) / den;
    const float sub2 = p * p * n2i + 2.0f * p * q * g + q * q * n2j;
    const float subn = fmaxf(sqrtf(fmaxf(sub2, 0.0f)), MIN_NORM);
    const float arg  = fminf(subn, TANH_CLIP);
    const float at   = 0.5f * logf((1.0f + arg) / (1.0f - arg));   // artanh
    const float s    = fmaxf(1.0f - n2i, MIN_NORM) * at / subn;    // 2/(sqrt_c*lam)
    const float w    = adj[i * NN + j] * dinv[i];

    Mh[i * NN + j] = (_Float16)(w * s * q);

    float wa = w * s * p;                       // reduce over j (lanes, same half)
#pragma unroll
    for (int mm = 8; mm >= 1; mm >>= 1) wa += __shfl_xor(wa, mm, 32);
    if (m == 0) atomicAdd(&rowA[i], wa);
  }
}

// ---- K2: agg = Mh @ x + diag(rowA) * x  (WMMA, K=1536) ----------------------
__global__ void k_agg(const _Float16* __restrict__ Mh,
                      const _Float16* __restrict__ xhT,
                      const float* __restrict__ rowA,
                      const float* __restrict__ x,
                      float* __restrict__ agg) {
  const int wave = (blockIdx.x * blockDim.x + threadIdx.x) >> 5;
  const int ti = wave >> 2;        // 0..95
  const int tn = wave & 3;         // 0..3 (64 cols / 16)
  const int i0 = ti * 16, n0 = tn * 16;
  const int lane = threadIdx.x & 31;
  const int m = lane & 15, half = lane >> 4;

  v8f d = {};
  for (int k0 = 0; k0 < NN; k0 += 32) {
    v16h a = load_a16(Mh, NN, i0, k0, m, half);
    v16h b = load_b16(xhT, NN, n0, k0, m, half);
    d = __builtin_amdgcn_wmma_f32_16x16x32_f16(false, a, false, b,
                                               (short)0, d, false, false);
  }
#pragma unroll
  for (int r = 0; r < 8; ++r) {
    const int i = i0 + r + 8 * half;
    const int c = n0 + m;
    agg[i * CC + c] = d[r] + rowA[i] * x[i * CC + c];
  }
}

// ---- K3: transformed = agg @ weight + bias; out = expmap_0(transformed) -----
// expmap at origin (c=1): tanh(||t||) * t / ||t||. One wave per row.
__global__ void k_out(const float* __restrict__ agg,
                      const float* __restrict__ weight,
                      const float* __restrict__ bias,
                      float* __restrict__ out) {
  const int i    = (blockIdx.x * blockDim.x + threadIdx.x) >> 5;
  const int lane = threadIdx.x & 31;
  const int c0 = lane, c1 = lane + 32;

  float t0 = bias[c0], t1 = bias[c1];
  for (int k = 0; k < CC; ++k) {
    const float ag = agg[i * CC + k];
    t0 = fmaf(ag, weight[k * CC + c0], t0);
    t1 = fmaf(ag, weight[k * CC + c1], t1);
  }
  float nsq = t0 * t0 + t1 * t1;
#pragma unroll
  for (int mm = 16; mm >= 1; mm >>= 1) nsq += __shfl_xor(nsq, mm, 32);
  const float nrm = fmaxf(sqrtf(nsq), MIN_NORM);
  const float sc  = tanhf(nrm) / nrm;
  out[i * CC + c0] = sc * t0;
  out[i * CC + c1] = sc * t1;
}

// ---- host launch ------------------------------------------------------------
extern "C" void kernel_launch(void* const* d_in, const int* in_sizes, int n_in,
                              void* d_out, int out_size, void* d_ws, size_t ws_size,
                              hipStream_t stream) {
  (void)in_sizes; (void)n_in; (void)out_size; (void)ws_size;
  const float* x      = (const float*)d_in[0];   // (1536, 64)
  const float* adj    = (const float*)d_in[1];   // (1536, 1536)
  const float* weight = (const float*)d_in[2];   // (64, 64)
  const float* bias   = (const float*)d_in[3];   // (64,)
  float* out = (float*)d_out;                    // (1536, 64)

  // workspace carve (all chunks are multiples of 32 bytes; base is 256B-aligned)
  char* base = (char*)d_ws;
  _Float16* xh  = (_Float16*)base; base += (size_t)NN * CC * 2;   // x as f16
  _Float16* xhT = (_Float16*)base; base += (size_t)CC * NN * 2;   // x^T as f16
  _Float16* Mh  = (_Float16*)base; base += (size_t)NN * NN * 2;   // W.*B as f16
  float* n2   = (float*)base; base += (size_t)NN * 4;             // ||x_i||^2
  float* dinv = (float*)base; base += (size_t)NN * 4;             // 1/deg
  float* rowA = (float*)base; base += (size_t)NN * 4;             // sum_j w*A
  float* agg  = (float*)base; base += (size_t)NN * CC * 4;        // tangent agg

  // 8 waves (wave32) per 256-thread block; every wave owns one row / tile.
  k_prep<<<dim3(NN / 8), dim3(256), 0, stream>>>(x, adj, xh, xhT, n2, dinv, rowA);
  k_pair<<<dim3(NT * NT / 8), dim3(256), 0, stream>>>(adj, xh, n2, dinv, Mh, rowA);
  k_agg <<<dim3(NT * (CC / 16) / 8), dim3(256), 0, stream>>>(Mh, xhT, rowA, x, agg);
  k_out <<<dim3(NN / 8), dim3(256), 0, stream>>>(agg, weight, bias, out);
}
